// AttentionModel_42013370089957
// MI455X (gfx1250) — matrix operand order repaired and checked by reading, not compiled, not run on hardware
//
#include <hip/hip_runtime.h>
#include <hip/hip_bf16.h>
#include <math.h>

// Problem constants (from reference setup_inputs)
constexpr int BB = 128;    // batch
constexpr int NN = 1000;   // nodes
constexpr int EE = 128;    // embed
constexpr int HH = 8;      // heads
constexpr int DD = 16;     // head dim
constexpr float SCALE = 0.25f;          // 1/sqrt(16)
constexpr float CLIP  = 10.0f;

typedef __attribute__((ext_vector_type(16))) __bf16 v16bf;
typedef __attribute__((ext_vector_type(8)))  float  v8f;

__device__ __forceinline__ float bflo(unsigned u) { return __uint_as_float(u << 16); }
__device__ __forceinline__ float bfhi(unsigned u) { return __uint_as_float(u & 0xffff0000u); }
__device__ __forceinline__ unsigned short f2bf(float f) {
    unsigned u = __float_as_uint(f);
    return (unsigned short)((u + 0x7fffu + ((u >> 16) & 1u)) >> 16);   // RNE
}
// expand 16 contiguous bf16 (32B, 16B-aligned) into 16 floats
__device__ __forceinline__ void bf16x16(const unsigned short* p, float* f) {
    uint4 u0 = *(const uint4*)p;
    uint4 u1 = *(const uint4*)(p + 8);
    unsigned w[8] = {u0.x, u0.y, u0.z, u0.w, u1.x, u1.y, u1.z, u1.w};
#pragma unroll
    for (int i = 0; i < 8; ++i) { f[2*i] = bflo(w[i]); f[2*i+1] = bfhi(w[i]); }
}

// ---------------- P0: weight prep (transpose + bf16 cast) ----------------
__global__ void prep_weights(const float* __restrict__ Wqkv, unsigned short* __restrict__ WqkvT,
                             const float* __restrict__ Wstep, float* __restrict__ WstepT,
                             const float* __restrict__ Wmlp,  float* __restrict__ WmlpT) {
    int tid = blockIdx.x * 256 + threadIdx.x;
    if (tid < 384 * 128) { int c = tid / 128, k = tid % 128; WqkvT[c*128 + k] = f2bf(Wqkv[k*384 + c]); }
    if (tid < 128 * 256) { int e = tid / 256, k = tid % 256; WstepT[e*256 + k] = Wstep[k*128 + e]; }
    if (tid < 128 * 128) { int e = tid / 128, k = tid % 128; WmlpT[e*128 + k]  = Wmlp[k*128 + e]; }
}

// ---------------- P2: fixed = graph_emb @ Wfix + bfix ----------------
__global__ void fixed_kernel(const float* __restrict__ ge, const float* __restrict__ Wfix,
                             const float* __restrict__ bfix, float* __restrict__ fixedB) {
    int tid = blockIdx.x * 128 + threadIdx.x;      // 128*128
    int b = tid >> 7, e = tid & 127;
    float acc = bfix[e];
    const float* g = ge + b * 128;
#pragma unroll 4
    for (int k = 0; k < 128; ++k) acc += g[k] * Wfix[k*128 + e];
    fixedB[tid] = acc;
}

// ---------------- P1: qkv GEMM via WMMA bf16 ----------------
// One wave per 16-row M tile; K=128 in 4 slices of 32; 24 column tiles.
// Outputs scattered to bf16 K[b][h][n][d], V[b][h][n][d], LK[b][n][e] (row-major).
__global__ __launch_bounds__(32) void qkv_gemm(
        const float* __restrict__ A,            // node_emb [B*N][128]
        const float* __restrict__ bias,         // bqkv[384]
        const unsigned short* __restrict__ Wt,  // [384][128] bf16 (pre-transposed)
        unsigned short* __restrict__ Kb,
        unsigned short* __restrict__ Vb,
        unsigned short* __restrict__ LKr) {
    int m0   = blockIdx.x * 16;
    int lane = threadIdx.x;
    int half = lane >> 4;
    int row  = m0 + (lane & 15);
    const float* ar = A + (size_t)row * 128;

    // A fragments for all 4 K-slices (lane<16 -> K 0..15, lane>=16 -> K 16..31 of slice)
    v16bf afr[4];
#pragma unroll
    for (int ks = 0; ks < 4; ++ks) {
        const float* p0 = ar + ks*32 + half*16;
        float4 f0 = *(const float4*)(p0);
        float4 f1 = *(const float4*)(p0 + 4);
        float4 f2 = *(const float4*)(p0 + 8);
        float4 f3 = *(const float4*)(p0 + 12);
        v16bf a;
        a[0]=(__bf16)f0.x; a[1]=(__bf16)f0.y; a[2]=(__bf16)f0.z; a[3]=(__bf16)f0.w;
        a[4]=(__bf16)f1.x; a[5]=(__bf16)f1.y; a[6]=(__bf16)f1.z; a[7]=(__bf16)f1.w;
        a[8]=(__bf16)f2.x; a[9]=(__bf16)f2.y; a[10]=(__bf16)f2.z; a[11]=(__bf16)f2.w;
        a[12]=(__bf16)f3.x; a[13]=(__bf16)f3.y; a[14]=(__bf16)f3.z; a[15]=(__bf16)f3.w;
        afr[ks] = a;
    }

    for (int c0 = 0; c0 < 384; c0 += 16) {
        int col = c0 + (lane & 15);
        v8f acc = {};
#pragma unroll
        for (int ks = 0; ks < 4; ++ks) {
            // B fragment: 16 contiguous K values per lane from transposed weights
            v16bf bf = *(const v16bf*)(Wt + (size_t)col*128 + ks*32 + half*16);
            acc = __builtin_amdgcn_wmma_f32_16x16x32_bf16(
                      false, afr[ks], false, bf, (short)0, acc, false, false);
        }
        float bv = bias[col];
#pragma unroll
        for (int r = 0; r < 8; ++r) {                     // C layout: lane=col, rows r(+8)
            int m = m0 + (half ? 8 : 0) + r;
            int b = m / NN, n = m % NN;
            unsigned short o = f2bf(acc[r] + bv);
            if (col < 128) {
                int h = col >> 4, dd = col & 15;
                Kb[(((size_t)(b*HH + h))*NN + n)*DD + dd] = o;
            } else if (col < 256) {
                int cc = col - 128, h = cc >> 4, dd = cc & 15;
                Vb[(((size_t)(b*HH + h))*NN + n)*DD + dd] = o;
            } else {
                int e = col - 256;
                LKr[((size_t)(b*NN + n))*EE + e] = o;      // row-major: wide per-row loads
            }
        }
    }
}

// ---------------- Persistent decode: one workgroup per batch element ----------------
// Visited nodes are *exactly* dead in the reference (score/logit forced to NEG,
// exp(NEG-max)==0 in fp32), so we keep a compacted sorted active-node list and
// never touch K/V/logit_k rows of visited nodes: ~2x less L2 streaming overall.
constexpr int TPB = 1024;   // 32 waves; head-group = 128 threads (4 waves) per head

__global__ __launch_bounds__(TPB) void decode_kernel(
        const float* __restrict__ nodeEmb, const float* __restrict__ fixedB,
        const float* __restrict__ WstepT,  const float* __restrict__ bstep,
        const float* __restrict__ WmlpT,   const float* __restrict__ bmlp,
        const unsigned short* __restrict__ Kb, const unsigned short* __restrict__ Vb,
        const unsigned short* __restrict__ LKr, float* __restrict__ out) {
    const int b    = blockIdx.x;
    const int tid  = threadIdx.x;
    const int lane = tid & 31;
    const int wid  = tid >> 5;
    const int hg   = tid >> 7;        // head group 0..7
    const int lt   = tid & 127;       // thread-in-head-group

    __shared__ __align__(16) float qS[EE];
    __shared__ __align__(16) float xS[EE];
    __shared__ __align__(16) float ctxS[EE];
    __shared__ float sS[HH][1024];    // per-head scores by list position
    __shared__ float lS[1024];        // logits by list position
    __shared__ int   listS[1024];     // sorted active node ids
    __shared__ float redS[32], red2[32];
    __shared__ int   redI[32];
    __shared__ int   curS, posS, cntS;
    __shared__ float totS, mS;

    if (tid < NN - 1) listS[tid] = tid + 1;            // node 0 pre-visited
    if (tid == 0) { curS = 0; totS = 0.f; cntS = NN - 1; }
    __syncthreads();

    const unsigned short* Kh  = Kb + ((size_t)(b*HH + hg))*NN*DD;
    const unsigned short* Vh  = Vb + ((size_t)(b*HH + hg))*NN*DD;
    const unsigned short* LKb = LKr + (size_t)b*NN*EE;
    const float* firstP = nodeEmb + (size_t)b*NN*EE;

    for (int t = 0; t < NN - 1; ++t) {
        const int cnt = cntS;          // active nodes this step (= NN-1-t)

        // (a) q = fixed + [first,last] @ Wstep + bstep  (b128 loads throughout)
        if (tid < EE) {
            const float4* w4 = (const float4*)(WstepT + tid*256);
            const float4* f4 = (const float4*)firstP;
            const float4* l4 = (const float4*)(nodeEmb + ((size_t)b*NN + curS)*EE);
            float acc = fixedB[b*EE + tid] + bstep[tid];
#pragma unroll 8
            for (int k = 0; k < 32; ++k) {
                float4 w = w4[k], f = f4[k];
                acc += f.x*w.x + f.y*w.y + f.z*w.z + f.w*w.w;
            }
#pragma unroll 8
            for (int k = 0; k < 32; ++k) {
                float4 w = w4[32 + k], l = l4[k];
                acc += l.x*w.x + l.y*w.y + l.z*w.z + l.w*w.w;
            }
            qS[tid] = acc;
        }
        __syncthreads();

        // (b) attention scores over active list + local max (scale folded into q)
        float qh[16];
#pragma unroll
        for (int j = 0; j < 16; ++j) qh[j] = qS[hg*16 + j] * SCALE;
        float ml = -3.0e38f;
        for (int i = lt; i < cnt; i += 128) {
            int n = listS[i];
            float kf[16]; bf16x16(Kh + (size_t)n*DD, kf);
            __builtin_prefetch(Vh + (size_t)n*DD, 0, 1);   // global_prefetch_b8 on V stream
            float sc = 0.f;
#pragma unroll
            for (int j = 0; j < 16; ++j) sc += kf[j] * qh[j];
            sS[hg][i] = sc;
            ml = fmaxf(ml, sc);
        }
        // (c) head max across 4 waves
#pragma unroll
        for (int off = 16; off >= 1; off >>= 1) ml = fmaxf(ml, __shfl_xor(ml, off, 32));
        if (lane == 0) redS[wid] = ml;
        __syncthreads();
        float hmax = fmaxf(fmaxf(redS[4*hg], redS[4*hg+1]), fmaxf(redS[4*hg+2], redS[4*hg+3]));

        // (d) exp + head sum; zero ctx
        float ls = 0.f;
        for (int i = lt; i < cnt; i += 128) {
            float p = __expf(sS[hg][i] - hmax);
            sS[hg][i] = p;
            ls += p;
        }
#pragma unroll
        for (int off = 16; off >= 1; off >>= 1) ls += __shfl_xor(ls, off, 32);
        if (lane == 0) red2[wid] = ls;
        if (tid < EE) ctxS[tid] = 0.f;
        __syncthreads();
        float inv = 1.f / (red2[4*hg] + red2[4*hg+1] + red2[4*hg+2] + red2[4*hg+3]);

        // (e) ctx = softmax @ V over active list (partials, shuffle, ds_add_f32)
        float fa[16];
#pragma unroll
        for (int j = 0; j < 16; ++j) fa[j] = 0.f;
        for (int i = lt; i < cnt; i += 128) {
            int n = listS[i];
            float p = sS[hg][i];
            float vf[16]; bf16x16(Vh + (size_t)n*DD, vf);
#pragma unroll
            for (int j = 0; j < 16; ++j) fa[j] += p * vf[j];
        }
#pragma unroll
        for (int off = 16; off >= 1; off >>= 1)
#pragma unroll
            for (int j = 0; j < 16; ++j) fa[j] += __shfl_xor(fa[j], off, 32);
        if (lane == 0)
#pragma unroll
            for (int j = 0; j < 16; ++j) atomicAdd(&ctxS[hg*16 + j], fa[j] * inv);
        __syncthreads();

        // (f) x = ctx @ Wmlp + bmlp  (b128 loads)
        if (tid < EE) {
            const float4* w4 = (const float4*)(WmlpT + tid*128);
            const float4* c4 = (const float4*)ctxS;
            float acc = bmlp[tid];
#pragma unroll 8
            for (int k = 0; k < 32; ++k) {
                float4 w = w4[k], c = c4[k];
                acc += c.x*w.x + c.y*w.y + c.z*w.z + c.w*w.w;
            }
            xS[tid] = acc;
        }
        __syncthreads();

        // (g) logits over active rows: 16 b128 global + 32 b128 DS loads/thread
        if (tid < cnt) {
            int n = listS[tid];
            const unsigned short* rp = LKb + (size_t)n * EE;   // 256B contiguous row
            float acc = 0.f;
#pragma unroll
            for (int ec = 0; ec < 8; ++ec) {
                float kf[16]; bf16x16(rp + ec*16, kf);
                const float4* x4 = (const float4*)(xS + ec*16);
                float4 x0 = x4[0], x1 = x4[1], x2 = x4[2], x3 = x4[3];
                acc += kf[0]*x0.x + kf[1]*x0.y + kf[2]*x0.z + kf[3]*x0.w;
                acc += kf[4]*x1.x + kf[5]*x1.y + kf[6]*x1.z + kf[7]*x1.w;
                acc += kf[8]*x2.x + kf[9]*x2.y + kf[10]*x2.z + kf[11]*x2.w;
                acc += kf[12]*x3.x + kf[13]*x3.y + kf[14]*x3.z + kf[15]*x3.w;
            }
            lS[tid] = tanhf(acc * SCALE) * CLIP;
        } else {
            lS[tid] = -3.0e38f;   // padding never wins argmax
        }
        __syncthreads();

        // (h) argmax over positions (list sorted ascending => first-index tiebreak ok)
        float v = lS[tid]; int ix = tid;
#pragma unroll
        for (int off = 16; off >= 1; off >>= 1) {
            float ov = __shfl_xor(v, off, 32); int oi = __shfl_xor(ix, off, 32);
            if (ov > v || (ov == v && oi < ix)) { v = ov; ix = oi; }
        }
        if (lane == 0) { redS[wid] = v; redI[wid] = ix; }
        __syncthreads();
        if (tid == 0) {
            float bv = redS[0]; int bi = redI[0];
            for (int w = 1; w < 32; ++w) if (redS[w] > bv) { bv = redS[w]; bi = redI[w]; }
            mS = bv; posS = bi; curS = listS[bi];
        }
        __syncthreads();
        // logsumexp over active positions
        float es = (tid < cnt) ? __expf(lS[tid] - mS) : 0.f;
#pragma unroll
        for (int off = 16; off >= 1; off >>= 1) es += __shfl_xor(es, off, 32);
        if (lane == 0) redS[wid] = es;
        __syncthreads();
        if (tid == 0) {
            float s = 0.f;
            for (int w = 0; w < 32; ++w) s += redS[w];
            totS += lS[posS] - mS - logf(s);
        }
        // compact the list: remove position posS (parallel left-shift)
        int myv = (tid < cnt) ? listS[tid] : 0;
        __syncthreads();
        if (tid < cnt && tid > posS) listS[tid - 1] = myv;
        if (tid == 0) cntS = cnt - 1;
        __syncthreads();
    }
    if (tid == 0) out[b] = totS;
}

// ---------------- host launcher ----------------
extern "C" void kernel_launch(void* const* d_in, const int* in_sizes, int n_in,
                              void* d_out, int out_size, void* d_ws, size_t ws_size,
                              hipStream_t stream) {
    const float* nodeEmb  = (const float*)d_in[0];
    const float* graphEmb = (const float*)d_in[1];
    const float* Wqkv     = (const float*)d_in[2];
    const float* bqkv     = (const float*)d_in[3];
    const float* Wfix     = (const float*)d_in[4];
    const float* bfix     = (const float*)d_in[5];
    const float* Wstep    = (const float*)d_in[6];
    const float* bstep    = (const float*)d_in[7];
    const float* Wmlp     = (const float*)d_in[8];
    const float* bmlp     = (const float*)d_in[9];
    float* out = (float*)d_out;

    char* ws = (char*)d_ws;
    size_t szKV     = (size_t)BB*HH*NN*DD*sizeof(unsigned short);   // 32.77 MB
    size_t offK     = 0;
    size_t offV     = offK + szKV;
    size_t offLK    = offV + szKV;
    size_t offWqkvT = offLK + (size_t)BB*NN*EE*sizeof(unsigned short);
    size_t offWstepT= offWqkvT + (size_t)384*128*sizeof(unsigned short);
    size_t offWmlpT = offWstepT + (size_t)128*256*sizeof(float);
    size_t offFixed = offWmlpT + (size_t)128*128*sizeof(float);
    // total ~98.6 MB of d_ws

    unsigned short* WqkvT = (unsigned short*)(ws + offWqkvT);
    float* WstepT = (float*)(ws + offWstepT);
    float* WmlpT  = (float*)(ws + offWmlpT);
    float* fixedB = (float*)(ws + offFixed);
    unsigned short* Kb  = (unsigned short*)(ws + offK);
    unsigned short* Vb  = (unsigned short*)(ws + offV);
    unsigned short* LKr = (unsigned short*)(ws + offLK);

    prep_weights<<<192, 256, 0, stream>>>(Wqkv, WqkvT, Wstep, WstepT, Wmlp, WmlpT);
    fixed_kernel<<<BB, 128, 0, stream>>>(graphEmb, Wfix, bfix, fixedB);
    qkv_gemm<<<(BB * NN) / 16, 32, 0, stream>>>(nodeEmb, bqkv, WqkvT, Kb, Vb, LKr);
    decode_kernel<<<BB, TPB, 0, stream>>>(nodeEmb, fixedB, WstepT, bstep, WmlpT, bmlp,
                                          Kb, Vb, LKr, out);
}
